// MultiSimilarityLoss_87230785781828
// MI455X (gfx1250) — matrix operand order repaired and checked by reading, not compiled, not run on hardware
//
#include <hip/hip_runtime.h>
#include <hip/hip_fp16.h>
#include <math.h>

typedef __attribute__((ext_vector_type(16))) _Float16 v16h;
typedef __attribute__((ext_vector_type(8)))  _Float16 v8h;
typedef __attribute__((ext_vector_type(8)))  float    v8f;

#define BN 4096      // batch (rows/cols of sim)
#define DK 256       // feature dim (K)
#define THRESH 0.5f
#define MARGIN 0.1f
#define SPOS 2.0f
#define SNEG 40.0f

// ---------------------------------------------------------------------------
// Phase 0: L2-normalize rows of feats, convert to f16 (WMMA operand matrix).
// One block per row, 256 threads = 8 waves.
// ---------------------------------------------------------------------------
__global__ void __launch_bounds__(256)
ms_norm_kernel(const float* __restrict__ feats, _Float16* __restrict__ nh) {
    const int row = blockIdx.x;
    const int t   = threadIdx.x;
    const float x = feats[row * DK + t];
    __shared__ float red[256];
    red[t] = x * x;
    __syncthreads();
    #pragma unroll
    for (int s = 128; s > 0; s >>= 1) {
        if (t < s) red[t] += red[t + s];
        __syncthreads();
    }
    const float inv = rsqrtf(red[0]);
    nh[row * DK + t] = (_Float16)(x * inv);
}

// ---------------------------------------------------------------------------
// WMMA fragment helpers (layouts per CDNA5 ISA 7.12.2, wave32).
// A 16x32 f16: lane m=lane&15 holds K = kb..kb+7 and kb+16..kb+23, kb=(lane>>4)*8
// B 32x16 f16: lane n=lane&15 holds K = kb..kb+7 and kb+16..kb+23, kb=(lane>>4)*8,
//              column n of B == row (colBase+n) of N  (since B = N^T block).
// C 16x16 f32: VGPR r, lanes 0-15 -> row r, lanes 16-31 -> row r+8; col=lane&15
// ---------------------------------------------------------------------------
__device__ __forceinline__ void ms_load_afrags(const _Float16* __restrict__ nh,
                                               int rowBase, int lane, v16h* a) {
    const int m  = lane & 15;
    const int kb = (lane >> 4) * 8;
    const _Float16* base = nh + (size_t)(rowBase + m) * DK;
    #pragma unroll
    for (int kc = 0; kc < 8; ++kc) {
        union { v16h v; v8h h[2]; } u;
        u.h[0] = *(const v8h*)(base + kc * 32 + kb);
        u.h[1] = *(const v8h*)(base + kc * 32 + 16 + kb);
        a[kc] = u.v;
    }
}

__device__ __forceinline__ v16h ms_load_bfrag(const _Float16* __restrict__ nh,
                                              int colBase, int kc, int lane) {
    const int n  = lane & 15;
    const int kb = (lane >> 4) * 8;
    const _Float16* p = nh + (size_t)(colBase + n) * DK + kc * 32 + kb;
    union { v16h v; v8h h[2]; } u;
    u.h[0] = *(const v8h*)(p);
    u.h[1] = *(const v8h*)(p + 16);
    return u.v;
}

__device__ __forceinline__ v8f ms_tile(const v16h* a, const _Float16* __restrict__ nh,
                                       int colBase, int lane) {
    v8f acc = {};
    #pragma unroll
    for (int kc = 0; kc < 8; ++kc) {
        v16h b = ms_load_bfrag(nh, colBase, kc, lane);
        acc = __builtin_amdgcn_wmma_f32_16x16x32_f16(
            /*neg_a=*/false, a[kc], /*neg_b=*/false, b,
            /*c_mod=*/(short)0, acc, /*reuse_a=*/false, /*reuse_b=*/false);
    }
    return acc;
}

// ---------------------------------------------------------------------------
// Phase 1 (mining): block = 16 rows x all 4096 cols; 4 waves, each wave takes
// every 4th 16-col tile. Per-row min(sim | pos) and max(sim | neg), finished
// entirely in-block (no atomics).
// ---------------------------------------------------------------------------
__global__ void __launch_bounds__(128)
ms_mine_kernel(const _Float16* __restrict__ nh, const int* __restrict__ labels,
               float* __restrict__ min_pos, float* __restrict__ max_neg) {
    const int rowBase = blockIdx.x * 16;
    const int lane = threadIdx.x & 31;
    const int wave = threadIdx.x >> 5;
    const int roff = (lane >> 4) * 8;   // rows r+0..7 (lanes 0-15) or r+8..15

    v16h a[8];
    ms_load_afrags(nh, rowBase, lane, a);

    int lblRow[8];
    #pragma unroll
    for (int r = 0; r < 8; ++r) lblRow[r] = labels[rowBase + roff + r];

    float mnp[8], mxn[8];
    #pragma unroll
    for (int r = 0; r < 8; ++r) { mnp[r] = INFINITY; mxn[r] = -INFINITY; }

    for (int tile = wave; tile < BN / 16; tile += 4) {
        const int colBase = tile * 16;
        v8f acc = ms_tile(a, nh, colBase, lane);
        const int lblCol = labels[colBase + (lane & 15)];
        #pragma unroll
        for (int r = 0; r < 8; ++r) {
            const float s = acc[r];
            if (lblCol == lblRow[r]) mnp[r] = fminf(mnp[r], s);
            else                     mxn[r] = fmaxf(mxn[r], s);
        }
    }

    // reduce across the 16-lane half-group (xor of bits 0..3 stays in-group)
    #pragma unroll
    for (int off = 8; off >= 1; off >>= 1) {
        #pragma unroll
        for (int r = 0; r < 8; ++r) {
            mnp[r] = fminf(mnp[r], __shfl_xor(mnp[r], off, 32));
            mxn[r] = fmaxf(mxn[r], __shfl_xor(mxn[r], off, 32));
        }
    }

    __shared__ float smn[4][16], smx[4][16];
    if ((lane & 15) == 0) {
        #pragma unroll
        for (int r = 0; r < 8; ++r) {
            smn[wave][roff + r] = mnp[r];
            smx[wave][roff + r] = mxn[r];
        }
    }
    __syncthreads();
    if (threadIdx.x < 16) {
        float m = INFINITY, x = -INFINITY;
        #pragma unroll
        for (int w = 0; w < 4; ++w) {
            m = fminf(m, smn[w][threadIdx.x]);
            x = fmaxf(x, smx[w][threadIdx.x]);
        }
        min_pos[rowBase + threadIdx.x] = m;
        max_neg[rowBase + threadIdx.x] = x;
    }
}

// ---------------------------------------------------------------------------
// Phase 2 (loss): recompute sim tiles (GEMM is ~free; N matrix is L2-resident),
// stream sim_weight once (the only large HBM traffic: 67 MB), accumulate
// masked exp-sums per row, emit per-row loss.
// ---------------------------------------------------------------------------
__global__ void __launch_bounds__(128)
ms_loss_kernel(const _Float16* __restrict__ nh, const int* __restrict__ labels,
               const float* __restrict__ sim_weight,
               const float* __restrict__ min_pos, const float* __restrict__ max_neg,
               float* __restrict__ rowloss) {
    const int rowBase = blockIdx.x * 16;
    const int lane = threadIdx.x & 31;
    const int wave = threadIdx.x >> 5;
    const int roff = (lane >> 4) * 8;

    v16h a[8];
    ms_load_afrags(nh, rowBase, lane, a);

    int lblRow[8];
    float mnp[8], mxn[8];
    bool  hneg[8];
    #pragma unroll
    for (int r = 0; r < 8; ++r) {
        const int row = rowBase + roff + r;
        lblRow[r] = labels[row];
        mnp[r]    = min_pos[row];
        mxn[r]    = max_neg[row];
        hneg[r]   = mxn[r] > -1e30f;
    }

    float psum[8], nsum[8];
    #pragma unroll
    for (int r = 0; r < 8; ++r) { psum[r] = 0.0f; nsum[r] = 0.0f; }

    for (int tile = wave; tile < BN / 16; tile += 4) {
        const int colBase = tile * 16;
        v8f acc = ms_tile(a, nh, colBase, lane);
        const int col = colBase + (lane & 15);
        const int lblCol = labels[col];
        const float* swp = sim_weight + (size_t)(rowBase + roff) * BN + col;
        #pragma unroll
        for (int r = 0; r < 8; ++r) {
            const float s  = acc[r];
            const float sw = swp[(size_t)r * BN];
            if (lblCol == lblRow[r]) {
                const bool keep = !hneg[r] || (s - MARGIN < mxn[r]);
                if (keep) psum[r] += expf(-SPOS * (s - THRESH) + sw);
            } else {
                if (s + MARGIN > mnp[r]) nsum[r] += expf(SNEG * (s - THRESH) - sw);
            }
        }
    }

    #pragma unroll
    for (int off = 8; off >= 1; off >>= 1) {
        #pragma unroll
        for (int r = 0; r < 8; ++r) {
            psum[r] += __shfl_xor(psum[r], off, 32);
            nsum[r] += __shfl_xor(nsum[r], off, 32);
        }
    }

    __shared__ float sps[4][16], sns[4][16];
    if ((lane & 15) == 0) {
        #pragma unroll
        for (int r = 0; r < 8; ++r) {
            sps[wave][roff + r] = psum[r];
            sns[wave][roff + r] = nsum[r];
        }
    }
    __syncthreads();
    if (threadIdx.x < 16) {
        float ps = 0.0f, ns = 0.0f;
        #pragma unroll
        for (int w = 0; w < 4; ++w) { ps += sps[w][threadIdx.x]; ns += sns[w][threadIdx.x]; }
        rowloss[rowBase + threadIdx.x] = log1pf(ps) / SPOS + log1pf(ns) / SNEG;
    }
}

// ---------------------------------------------------------------------------
// Phase 3: deterministic fixed-order reduction of 4096 row losses -> mean.
// ---------------------------------------------------------------------------
__global__ void __launch_bounds__(256)
ms_reduce_kernel(const float* __restrict__ rowloss, float* __restrict__ out) {
    __shared__ float red[256];
    const int t = threadIdx.x;
    float s = 0.0f;
    for (int i = t; i < BN; i += 256) s += rowloss[i];
    red[t] = s;
    __syncthreads();
    #pragma unroll
    for (int st = 128; st > 0; st >>= 1) {
        if (t < st) red[t] += red[t + st];
        __syncthreads();
    }
    if (t == 0) out[0] = red[0] / (float)BN;
}

extern "C" void kernel_launch(void* const* d_in, const int* in_sizes, int n_in,
                              void* d_out, int out_size, void* d_ws, size_t ws_size,
                              hipStream_t stream) {
    const float* feats      = (const float*)d_in[0];
    const float* sim_weight = (const float*)d_in[1];
    const int*   labels     = (const int*)d_in[2];

    char* ws = (char*)d_ws;
    _Float16* nh      = (_Float16*)ws;                          // 4096*256*2  = 2 MiB
    float*    min_pos = (float*)(ws + (size_t)2 * 1024 * 1024); // 16 KB
    float*    max_neg = (float*)(ws + (size_t)2 * 1024 * 1024 + 16384);
    float*    rowloss = (float*)(ws + (size_t)2 * 1024 * 1024 + 32768);

    ms_norm_kernel  <<<BN,      256, 0, stream>>>(feats, nh);
    ms_mine_kernel  <<<BN / 16, 128, 0, stream>>>(nh, labels, min_pos, max_neg);
    ms_loss_kernel  <<<BN / 16, 128, 0, stream>>>(nh, labels, sim_weight,
                                                  min_pos, max_neg, rowloss);
    ms_reduce_kernel<<<1,       256, 0, stream>>>(rowloss, (float*)d_out);
}
